// EGAT_78151224918214
// MI455X (gfx1250) — matrix operand order repaired and checked by reading, not compiled, not run on hardware
//
#include <hip/hip_runtime.h>
#include <hip/hip_bf16.h>

// ---------------- problem constants (from reference) ----------------
#define NG 76800
#define EG 307200
#define NL 38400
#define EL 153600
#define NB 64
#define NH 8
#define BOND 40
#define ATOM 92
#define IN_E 10
#define FE1  36

// padded K dims (multiple of 32 for v_wmma_f32_16x16x32_f16)
#define K1N 64   // 40  -> 64
#define K1E 32   // 10  -> 32
#define K2N 96   // 92  -> 96
#define KEF 64   // 40  -> 64

typedef __attribute__((ext_vector_type(16))) _Float16 v16h;
typedef __attribute__((ext_vector_type(8)))  _Float16 v8h;
typedef __attribute__((ext_vector_type(8)))  float    v8f;

// =====================================================================
// Weight pre-pack: f32 [K,N] row-major -> f16 fragment-order
//   out[((ct*KB + kb)*32 + lane)*16 + i] = W[kb*32 + (lane>>4)*16 + i][ct*16 + (lane&15)]
// so each lane's B fragment is one contiguous 32-byte chunk.
// =====================================================================
__global__ void pack_w_f16(const float* __restrict__ W, _Float16* __restrict__ out,
                           int K, int N, int KB, int CT)
{
    long idx = blockIdx.x * (long)blockDim.x + threadIdx.x;
    long total = (long)CT * KB * 512;
    if (idx >= total) return;
    int i    = (int)(idx & 15);
    int lane = (int)((idx >> 4) & 31);
    long blk = idx >> 9;
    int kb   = (int)(blk % KB);
    int ct   = (int)(blk / KB);
    int k = kb * 32 + (lane >> 4) * 16 + i;
    int c = ct * 16 + (lane & 15);
    out[idx] = (k < K && c < N) ? (_Float16)W[(size_t)k * N + c] : (_Float16)0.f;
}

// =====================================================================
// WMMA GEMM:  C[M, ldc] (f32, at column offset coff) = A[M,lda](f16) * W
// Wp = fragment-packed weights.  KB = K/32, TN = column tiles per wave
// (both templated -> fully unrolled straight-line body, no branches).
// Each wave computes a 16 x (16*TN) strip; one A fragment feeds TN WMMAs
// per K step.  Exact tiling: grid = (M/64, N/(16*TN)).
// =====================================================================
template<int KB, int TN>
__global__ void wmma_gemm_kernel(const _Float16* __restrict__ A, int lda,
                                 const _Float16* __restrict__ Wp,
                                 float* __restrict__ C, int ldc, int coff)
{
    const int wave = threadIdx.x >> 5;
    const int lane = threadIdx.x & 31;
    const int half = lane >> 4;
    const int r    = lane & 15;
    const int row0 = (blockIdx.x * 4 + wave) * 16;
    const int ct0  = blockIdx.y * TN;      // first column tile of this wave

    v8f acc[TN];
#pragma unroll
    for (int t = 0; t < TN; ++t) acc[t] = (v8f){};

#pragma unroll
    for (int kb = 0; kb < KB; ++kb) {
        // A fragment: lanes 0-15 rows, K kb*32 + {0-15}; lanes 16-31 {16-31}
        const _Float16* arow = A + (size_t)(row0 + r) * lda + kb * 32 + half * 16;
        v8h alo = *(const v8h*)arow;
        v8h ahi = *(const v8h*)(arow + 8);
        v16h a;
#pragma unroll
        for (int i = 0; i < 8; ++i) { a[i] = alo[i]; a[8 + i] = ahi[i]; }
#pragma unroll
        for (int t = 0; t < TN; ++t) {
            const v8h* wp = (const v8h*)(Wp +
                ((size_t)((ct0 + t) * KB + kb) * 32 + lane) * 16);
            v8h blo = wp[0];
            v8h bhi = wp[1];
            v16h b;
#pragma unroll
            for (int i = 0; i < 8; ++i) { b[i] = blo[i]; b[8 + i] = bhi[i]; }
            acc[t] = __builtin_amdgcn_wmma_f32_16x16x32_f16(
                         false, a, false, b, (short)0, acc[t], false, false);
        }
    }
    // C/D layout: VGPR i -> row = i + 8*half, col = lane&15
#pragma unroll
    for (int t = 0; t < TN; ++t) {
        float* cbase = C + (size_t)(row0 + half * 8) * ldc + coff + (ct0 + t) * 16 + r;
#pragma unroll
        for (int i = 0; i < 8; ++i) cbase[(size_t)i * ldc] = acc[t][i];
    }
}

// =====================================================================
// helpers
// =====================================================================
__global__ void fill_kernel(float* __restrict__ p, float v, long n)
{
    long i = blockIdx.x * (long)blockDim.x + threadIdx.x;
    if (i < n) p[i] = v;
}

// f32 [R,C] row-major -> f16 [Rp, ldo] zero-padded
__global__ void cvt_pad_f16(const float* __restrict__ in, _Float16* __restrict__ out,
                            int R, int C, int Rp, int ldo)
{
    long i = blockIdx.x * (long)blockDim.x + threadIdx.x;
    long total = (long)Rp * ldo;
    if (i >= total) return;
    int r = (int)(i / ldo), c = (int)(i % ldo);
    float v = (r < R && c < C) ? in[(size_t)r * C + c] : 0.f;
    out[i] = (_Float16)v;
}

// ef = repeat(y2n, 2, axis=0) + lg_efeat, emitted directly as padded f16
__global__ void build_ef_f16(const float* __restrict__ y2n, const float* __restrict__ lg_ef,
                             _Float16* __restrict__ out, int E, int C, int ldo)
{
    long i = blockIdx.x * (long)blockDim.x + threadIdx.x;
    long total = (long)E * ldo;
    if (i >= total) return;
    int e = (int)(i / ldo), c = (int)(i % ldo);
    float v = (c < C) ? (y2n[(size_t)(e >> 1) * C + c] + lg_ef[(size_t)e * C + c]) : 0.f;
    out[i] = (_Float16)v;
}

__device__ inline void atomicMaxF(float* addr, float val)
{
    unsigned int* u = (unsigned int*)addr;
    unsigned int old = *u;
    while (__uint_as_float(old) < val) {
        unsigned int assumed = old;
        old = atomicCAS(u, assumed, __float_as_uint(val));
        if (old == assumed) break;
    }
}

// =====================================================================
// Edge pass 1: f_edge = F + P_ni[src] + P_nj[dst] + bias ; leaky_relu;
// per-head attention scores; head-summed f_out; segment max via atomics.
// =====================================================================
template<int FE>
__global__ void edge_score_kernel(const float* __restrict__ F,      // [E, NH*FE]
                                  const float* __restrict__ P, int ldp,
                                  const int* __restrict__ src, const int* __restrict__ dst,
                                  const float* __restrict__ bias,   // [NH*FE]
                                  const float* __restrict__ attn,   // [NH, FE]
                                  float* __restrict__ scores,       // [E, NH]
                                  float* __restrict__ segmax,       // [N, NH]
                                  float* __restrict__ esum, int esum_limit,
                                  int E)
{
    int e = blockIdx.x * blockDim.x + threadIdx.x;
    if (e >= E) return;
    const int s = src[e], d = dst[e];
    const float* frow = F + (size_t)e * (NH * FE);
    const float* pni  = P + (size_t)s * ldp;              // ni block at col 0
    const float* pnj  = P + (size_t)d * ldp + NH * FE;    // nj block at col NH*FE
    float sc[NH];
#pragma unroll
    for (int h = 0; h < NH; ++h) sc[h] = 0.f;
    const bool we = (esum != nullptr) && (e < esum_limit);
    for (int f = 0; f < FE; ++f) {
        float fs = 0.f;
#pragma unroll
        for (int h = 0; h < NH; ++h) {
            const int j = h * FE + f;
            float v = frow[j] + pni[j] + pnj[j] + bias[j];
            v = fmaxf(v, 0.01f * v);          // leaky_relu, slope 0.01
            sc[h] += v * attn[j];
            fs    += v;
        }
        if (we) esum[(size_t)e * FE + f] = fs;
    }
#pragma unroll
    for (int h = 0; h < NH; ++h) {
        scores[(size_t)e * NH + h] = sc[h];
        atomicMaxF(&segmax[(size_t)d * NH + h], sc[h]);
    }
}

// Edge pass 2: ex = exp(score - max[dst]); accumulate segment sum
__global__ void softmax_num_kernel(const int* __restrict__ dst, float* __restrict__ scores,
                                   const float* __restrict__ segmax, float* __restrict__ segsum,
                                   long EH)
{
    long i = blockIdx.x * (long)blockDim.x + threadIdx.x;
    if (i >= EH) return;
    int e = (int)(i >> 3), h = (int)(i & 7);
    int d = dst[e];
    float ex = __expf(scores[i] - segmax[(size_t)d * NH + h]);
    scores[i] = ex;
    atomicAdd(&segsum[(size_t)d * NH + h], ex);
}

// Edge pass 3: head-summed weighted aggregation of (node proj + b_node)
template<int FN>
__global__ void aggregate_kernel(const float* __restrict__ P, int ldp, int nodeoff,
                                 const float* __restrict__ b_node,
                                 const int* __restrict__ src, const int* __restrict__ dst,
                                 const float* __restrict__ scores, const float* __restrict__ segsum,
                                 float* __restrict__ out, int E)
{
    int e = blockIdx.x * blockDim.x + threadIdx.x;
    if (e >= E) return;
    const int s = src[e], d = dst[e];
    float a[NH];
#pragma unroll
    for (int h = 0; h < NH; ++h)
        a[h] = scores[(size_t)e * NH + h] / segsum[(size_t)d * NH + h];
    const float* pn = P + (size_t)s * ldp + nodeoff;
    float* orow = out + (size_t)d * FN;
    for (int f = 0; f < FN; ++f) {
        float v = 0.f;
#pragma unroll
        for (int h = 0; h < NH; ++h) v += (pn[h * FN + f] + b_node[h * FN + f]) * a[h];
        atomicAdd(&orow[f], v);
    }
}

// =====================================================================
// pooling + readout
// =====================================================================
__global__ void pool_mean_kernel(const float* __restrict__ in, int rowsPerB, int C,
                                 float* __restrict__ ycat, int coff)
{
    int b = blockIdx.x;
    int f = threadIdx.x;
    if (f >= C) return;
    const float* base = in + (size_t)b * rowsPerB * C + f;
    float s = 0.f;
    for (int r = 0; r < rowsPerB; ++r) s += base[(size_t)r * C];
    ycat[b * 128 + coff + f] = s / (float)rowsPerB;
}

__global__ void readout_kernel(const float* __restrict__ ycat,
                               const float* __restrict__ Wr1, const float* __restrict__ br1,
                               const float* __restrict__ Wr2, const float* __restrict__ br2,
                               float* __restrict__ out)
{
    __shared__ float hbuf[NB * 128];
    int t = threadIdx.x;   // 128 threads
    for (int b = 0; b < NB; ++b) {
        float s = br1[t];
        for (int k = 0; k < 128; ++k) s += ycat[b * 128 + k] * Wr1[k * 128 + t];
        hbuf[b * 128 + t] = 1.f / (1.f + __expf(-s));
    }
    __syncthreads();
    if (t < NB) {
        float s = br2[0];
        for (int k = 0; k < 128; ++k) s += hbuf[t * 128 + k] * Wr2[k];
        out[t] = 1.f / (1.f + __expf(-s));
    }
}

// =====================================================================
// launcher
// =====================================================================
extern "C" void kernel_launch(void* const* d_in, const int* in_sizes, int n_in,
                              void* d_out, int out_size, void* d_ws, size_t ws_size,
                              hipStream_t stream)
{
    (void)in_sizes; (void)n_in; (void)out_size; (void)ws_size;

    const int*   gg_src   = (const int*)  d_in[0];
    const int*   gg_dst   = (const int*)  d_in[1];
    const float* gg_nfeat = (const float*)d_in[2];
    const float* gg_efeat = (const float*)d_in[3];
    const int*   lg_src   = (const int*)  d_in[4];
    const int*   lg_dst   = (const int*)  d_in[5];
    const float* lg_nfeat = (const float*)d_in[6];
    const float* lg_efeat = (const float*)d_in[7];
    const float* W_node1  = (const float*)d_in[8];
    const float* b_node1  = (const float*)d_in[9];
    const float* W_ni1    = (const float*)d_in[10];
    const float* W_nj1    = (const float*)d_in[11];
    const float* W_fij1   = (const float*)d_in[12];
    const float* attn1    = (const float*)d_in[13];
    const float* bias1    = (const float*)d_in[14];
    const float* W_node2a = (const float*)d_in[15];
    const float* b_node2a = (const float*)d_in[16];
    const float* W_ni2a   = (const float*)d_in[17];
    const float* W_nj2a   = (const float*)d_in[18];
    const float* W_fij2a  = (const float*)d_in[19];
    const float* attn2a   = (const float*)d_in[20];
    const float* bias2a   = (const float*)d_in[21];
    const float* W_node2b = (const float*)d_in[22];
    const float* b_node2b = (const float*)d_in[23];
    const float* W_ni2b   = (const float*)d_in[24];
    const float* W_nj2b   = (const float*)d_in[25];
    const float* W_fij2b  = (const float*)d_in[26];
    const float* attn2b   = (const float*)d_in[27];
    const float* bias2b   = (const float*)d_in[28];
    const float* Wr1      = (const float*)d_in[29];
    const float* br1      = (const float*)d_in[30];
    const float* Wr2      = (const float*)d_in[31];
    const float* br2      = (const float*)d_in[32];
    float* out = (float*)d_out;

    // ---------------- workspace bump allocator ----------------
    char* wbase = (char*)d_ws;
    size_t off = 0;
    auto alloc = [&](size_t bytes) -> void* {
        void* p = wbase + off;
        off = (off + bytes + 255) & ~(size_t)255;
        return p;
    };
    auto walloc = [&](int N, int KB) -> _Float16* {   // packed weight buffer
        return (_Float16*)alloc((size_t)(N / 16) * KB * 512 * 2);
    };

    _Float16* A1n   = (_Float16*)alloc((size_t)NG * K1N * 2);
    _Float16* A1e   = (_Float16*)alloc((size_t)EG * K1E * 2);
    _Float16* A2n   = (_Float16*)alloc((size_t)NL * K2N * 2);
    _Float16* Aef   = (_Float16*)alloc((size_t)EL * KEF * 2);
    _Float16* A2nb  = (_Float16*)alloc((size_t)NL * K2N * 2);
    _Float16* wp_ni1   = walloc(288, 2);
    _Float16* wp_nj1   = walloc(288, 2);
    _Float16* wp_node1 = walloc(320, 2);
    _Float16* wp_fij1  = walloc(288, 1);
    _Float16* wp_ni2a   = walloc(320, 3);
    _Float16* wp_nj2a   = walloc(320, 3);
    _Float16* wp_node2a = walloc(736, 3);
    _Float16* wp_fij2a  = walloc(320, 2);
    _Float16* wp_ni2b   = walloc(320, 3);
    _Float16* wp_nj2b   = walloc(320, 3);
    _Float16* wp_node2b = walloc(736, 3);
    _Float16* wp_fij2b  = walloc(320, 2);

    size_t Pfloats = (size_t)NG * 896;            // > NL*1376
    size_t Ffloats = (size_t)EG * 288;            // > EL*320
    float* P      = (float*)alloc(Pfloats * 4);
    float* F      = (float*)alloc(Ffloats * 4);
    float* scores = (float*)alloc((size_t)EG * NH * 4);
    float* segmax = (float*)alloc((size_t)NG * NH * 4);
    float* segsum = (float*)alloc((size_t)NG * NH * 4);
    float* y2n    = (float*)alloc((size_t)NG * BOND * 4);
    float* e1sum  = (float*)alloc((size_t)NG * FE1 * 4);
    float* e2sum  = (float*)alloc((size_t)EL * BOND * 4);
    float* nf2a   = (float*)alloc((size_t)NL * ATOM * 4);
    float* nf2b   = (float*)alloc((size_t)NL * ATOM * 4);
    float* ycat   = (float*)alloc((size_t)NB * 128 * 4);

    auto cvt = [&](const float* in, _Float16* o, int R, int C, int Rp, int ldo) {
        long total = (long)Rp * ldo;
        cvt_pad_f16<<<dim3((unsigned)((total + 255) / 256)), dim3(256), 0, stream>>>(in, o, R, C, Rp, ldo);
    };
    auto packw = [&](const float* W, _Float16* o, int K, int N, int KB) {
        int CT = N / 16;
        long total = (long)CT * KB * 512;
        pack_w_f16<<<dim3((unsigned)((total + 255) / 256)), dim3(256), 0, stream>>>(W, o, K, N, KB, CT);
    };
    // TN=4 when N%64==0, else TN=2 (all N here are multiples of 32) -> no guards
    auto gemm = [&](const _Float16* A, int lda, const _Float16* Wp, int KB,
                    float* C, int ldc, int coff, int M, int N) {
        dim3 b(128);
        if (N % 64 == 0) {
            dim3 g(M / 64, N / 64);
            switch (KB) {
                case 1: wmma_gemm_kernel<1, 4><<<g, b, 0, stream>>>(A, lda, Wp, C, ldc, coff); break;
                case 2: wmma_gemm_kernel<2, 4><<<g, b, 0, stream>>>(A, lda, Wp, C, ldc, coff); break;
                default: wmma_gemm_kernel<3, 4><<<g, b, 0, stream>>>(A, lda, Wp, C, ldc, coff); break;
            }
        } else {
            dim3 g(M / 64, N / 32);
            switch (KB) {
                case 1: wmma_gemm_kernel<1, 2><<<g, b, 0, stream>>>(A, lda, Wp, C, ldc, coff); break;
                case 2: wmma_gemm_kernel<2, 2><<<g, b, 0, stream>>>(A, lda, Wp, C, ldc, coff); break;
                default: wmma_gemm_kernel<3, 2><<<g, b, 0, stream>>>(A, lda, Wp, C, ldc, coff); break;
            }
        }
    };
    auto fill = [&](float* p, float v, long n) {
        fill_kernel<<<dim3((unsigned)((n + 255) / 256)), dim3(256), 0, stream>>>(p, v, n);
    };

    // ---------------- f16 conversions / weight packing ----------------
    cvt(gg_nfeat, A1n, NG, BOND, NG, K1N);
    cvt(gg_efeat, A1e, EG, IN_E, EG, K1E);
    cvt(lg_nfeat, A2n, NL, ATOM, NL, K2N);
    packw(W_ni1,   wp_ni1,   BOND, 288, 2);
    packw(W_nj1,   wp_nj1,   BOND, 288, 2);
    packw(W_node1, wp_node1, BOND, 320, 2);
    packw(W_fij1,  wp_fij1,  IN_E, 288, 1);
    packw(W_ni2a,   wp_ni2a,   ATOM, 320, 3);
    packw(W_nj2a,   wp_nj2a,   ATOM, 320, 3);
    packw(W_node2a, wp_node2a, ATOM, 736, 3);
    packw(W_fij2a,  wp_fij2a,  BOND, 320, 2);
    packw(W_ni2b,   wp_ni2b,   ATOM, 320, 3);
    packw(W_nj2b,   wp_nj2b,   ATOM, 320, 3);
    packw(W_node2b, wp_node2b, ATOM, 736, 3);
    packw(W_fij2b,  wp_fij2b,  BOND, 320, 2);

    // ================= stage 1: EGAT on gg =================
    fill(segmax, -3.0e38f, (long)NG * NH);
    fill(segsum, 0.f,      (long)NG * NH);
    fill(y2n,    0.f,      (long)NG * BOND);

    gemm(A1n, K1N, wp_ni1,   2, P, 896, 0,   NG, 288);
    gemm(A1n, K1N, wp_nj1,   2, P, 896, 288, NG, 288);
    gemm(A1n, K1N, wp_node1, 2, P, 896, 576, NG, 320);
    gemm(A1e, K1E, wp_fij1,  1, F, 288, 0,   EG, 288);

    edge_score_kernel<FE1><<<dim3((EG + 255) / 256), dim3(256), 0, stream>>>(
        F, P, 896, gg_src, gg_dst, bias1, attn1, scores, segmax, e1sum, NG, EG);
    softmax_num_kernel<<<dim3((EG * NH + 255) / 256), dim3(256), 0, stream>>>(
        gg_dst, scores, segmax, segsum, (long)EG * NH);
    aggregate_kernel<BOND><<<dim3((EG + 255) / 256), dim3(256), 0, stream>>>(
        P, 896, 576, b_node1, gg_src, gg_dst, scores, segsum, y2n, EG);

    // ================= stage 2a: EGAT on lg =================
    build_ef_f16<<<dim3((unsigned)(((long)EL * KEF + 255) / 256)), dim3(256), 0, stream>>>(
        y2n, lg_efeat, Aef, EL, BOND, KEF);
    fill(segmax, -3.0e38f, (long)NL * NH);
    fill(segsum, 0.f,      (long)NL * NH);
    fill(nf2a,   0.f,      (long)NL * ATOM);

    gemm(A2n, K2N, wp_ni2a,   3, P, 1376, 0,   NL, 320);
    gemm(A2n, K2N, wp_nj2a,   3, P, 1376, 320, NL, 320);
    gemm(A2n, K2N, wp_node2a, 3, P, 1376, 640, NL, 736);
    gemm(Aef, KEF, wp_fij2a,  2, F, 320,  0,   EL, 320);

    edge_score_kernel<BOND><<<dim3((EL + 255) / 256), dim3(256), 0, stream>>>(
        F, P, 1376, lg_src, lg_dst, bias2a, attn2a, scores, segmax, e2sum, EL, EL);
    softmax_num_kernel<<<dim3((EL * NH + 255) / 256), dim3(256), 0, stream>>>(
        lg_dst, scores, segmax, segsum, (long)EL * NH);
    aggregate_kernel<ATOM><<<dim3((EL + 255) / 256), dim3(256), 0, stream>>>(
        P, 1376, 640, b_node2a, lg_src, lg_dst, scores, segsum, nf2a, EL);

    // ================= stage 2b: EGAT on lg =================
    cvt(nf2a,  A2nb, NL, ATOM, NL, K2N);
    cvt(e2sum, Aef,  EL, BOND, EL, KEF);
    fill(segmax, -3.0e38f, (long)NL * NH);
    fill(segsum, 0.f,      (long)NL * NH);
    fill(nf2b,   0.f,      (long)NL * ATOM);

    gemm(A2nb, K2N, wp_ni2b,   3, P, 1376, 0,   NL, 320);
    gemm(A2nb, K2N, wp_nj2b,   3, P, 1376, 320, NL, 320);
    gemm(A2nb, K2N, wp_node2b, 3, P, 1376, 640, NL, 736);
    gemm(Aef,  KEF, wp_fij2b,  2, F, 320,  0,   EL, 320);

    edge_score_kernel<BOND><<<dim3((EL + 255) / 256), dim3(256), 0, stream>>>(
        F, P, 1376, lg_src, lg_dst, bias2b, attn2b, scores, segmax, nullptr, 0, EL);
    softmax_num_kernel<<<dim3((EL * NH + 255) / 256), dim3(256), 0, stream>>>(
        lg_dst, scores, segmax, segsum, (long)EL * NH);
    aggregate_kernel<ATOM><<<dim3((EL + 255) / 256), dim3(256), 0, stream>>>(
        P, 1376, 640, b_node2b, lg_src, lg_dst, scores, segsum, nf2b, EL);

    // ================= pooling + readout =================
    pool_mean_kernel<<<dim3(NB), dim3(128), 0, stream>>>(nf2b, NL / NB, ATOM, ycat, 0);
    pool_mean_kernel<<<dim3(NB), dim3(64),  0, stream>>>(e1sum, NG / NB, FE1, ycat, ATOM);
    readout_kernel<<<dim3(1), dim3(128), 0, stream>>>(ycat, Wr1, br1, Wr2, br2, out);
}
